// CharSeq2Seq_40372692582789
// MI455X (gfx1250) — compile-verified, matrix operand order checked by reading, not run on hardware
//
#include <hip/hip_runtime.h>
#include <hip/hip_bf16.h>

typedef __attribute__((ext_vector_type(16))) __bf16 v16bf;
typedef __attribute__((ext_vector_type(8)))  float  v8f;
typedef unsigned int u32;
typedef __attribute__((ext_vector_type(4)))  u32    u32x4;

namespace {
constexpr int kB   = 2048;
constexpr int kS   = 24;
constexpr int kT   = 24;
constexpr int kV   = 37;
constexpr int kE   = 256;
constexpr int kH   = 1024;
constexpr int k4H  = 4096;
constexpr int kSOS = 1;
constexpr int LDSTR = 40;                    // padded LDS k-stride (elements); 80B rows
constexpr u32 kBufBytes = 128 * LDSTR * 2;   // 10240B per tile buffer
}

union Frag { v16bf v; u32x4 u[2]; };

__device__ __forceinline__ float sigf(float x) { return 1.0f / (1.0f + __expf(-x)); }

// CDNA5 async global->LDS (ASYNCcnt-tracked, per-lane LDS offset in VGPR).
__device__ __forceinline__ void async_g2l_b128(u32 lds_off, const void* gaddr) {
  asm volatile("global_load_async_to_lds_b128 %0, %1, off"
               :: "v"(lds_off), "v"(gaddr)
               : "memory");
}
__device__ __forceinline__ void wait_async0() {
#if __has_builtin(__builtin_amdgcn_s_wait_asynccnt)
  __builtin_amdgcn_s_wait_asynccnt(0);
#else
  asm volatile("s_wait_asynccnt 0" ::: "memory");
#endif
}

// ---------------------------------------------------------------------------
// gates[m,n] = sum_k A[m,k] * Wcat[n,k] + bias[n]
// A = [seg0 | seg1] bf16; seg0 optionally gathered by token id (embedding rows)
// M=2048, N=4096, K=K0+K1 (1280 or 2048). 128x128 tile, 8 waves/WG, each wave
// owns 64x32 of C via 4x2 v_wmma_f32_16x16x32_bf16 per 32-deep K-step.
// Double-buffered LDS fed by global_load_async_to_lds_b128; async loads for
// tile i+1 overlap the 8 WMMAs of tile i.
// ---------------------------------------------------------------------------
__global__ __launch_bounds__(256)
void gates_gemm(const __bf16* __restrict__ seg0, int lda0, int K0,
                const __bf16* __restrict__ seg1, int K1,
                int gather, const int* __restrict__ tok, int tok_stride, int fixed_tok,
                const __bf16* __restrict__ Wcat,   // [4096, K0+K1] row-major
                const float*  __restrict__ bias,   // [4096]
                float*        __restrict__ outg)   // [2048, 4096]
{
  __shared__ __bf16 As[2][128 * LDSTR];
  __shared__ __bf16 Bs[2][128 * LDSTR];

  const int tid  = threadIdx.x;
  const int lane = tid & 31;
  const int wave = tid >> 5;
  const int wm   = wave >> 2;        // 0..1 : 64-row slab
  const int wn   = wave & 3;         // 0..3 : 32-col slab
  const int bm   = blockIdx.y * 128;
  const int bn   = blockIdx.x * 128;
  const int Kt   = K0 + K1;
  const int niter = Kt >> 5;

  // ---- hoisted per-thread staging state: each thread always services the
  //      same two (row, kc) chunks of the 128x32 A and B tiles ----
  const __bf16* a0p[2];   // seg0 base (token-gathered row for layer-0)
  const __bf16* a1p[2];   // seg1 base
  const __bf16* bwp[2];   // weight row base
  u32 lofA[2], lofB[2];   // LDS byte offsets (buffer 0)
#pragma unroll
  for (int i = 0; i < 2; ++i) {
    const int c  = tid + i * 256;    // chunk 0..511 (16B each)
    const int r  = c >> 2;           // tile row 0..127
    const int kc = (c & 3) * 8;      // k offset within tile
    int row0;
    if (gather) row0 = tok ? tok[(bm + r) * tok_stride] : fixed_tok;
    else        row0 = bm + r;
    a0p[i]  = seg0 + (long)row0 * lda0 + kc;
    a1p[i]  = seg1 + (long)(bm + r) * kH + kc;
    bwp[i]  = Wcat + (long)(bn + r) * Kt + kc;
    lofA[i] = (u32)(size_t)&As[0][r * LDSTR + kc];
    lofB[i] = (u32)(size_t)&Bs[0][r * LDSTR + kc];
  }

  auto issue_tile = [&](int it) {
    const int k0 = it << 5;
    const u32 bo = (it & 1) ? kBufBytes : 0u;
    const bool in0 = (k0 < K0);        // uniform: k-tiles never straddle K0
#pragma unroll
    for (int i = 0; i < 2; ++i) {
      const __bf16* ap = in0 ? (a0p[i] + k0) : (a1p[i] + (k0 - K0));
      async_g2l_b128(lofA[i] + bo, ap);
      async_g2l_b128(lofB[i] + bo, bwp[i] + k0);
    }
  };

  v8f zero = {};
  v8f acc[4][2];
#pragma unroll
  for (int mi = 0; mi < 4; ++mi)
#pragma unroll
    for (int ni = 0; ni < 2; ++ni) acc[mi][ni] = zero;

  const int lrow = lane & 15;
  const int koff = (lane < 16) ? 0 : 8;

  issue_tile(0);
  for (int it = 0; it < niter; ++it) {
    wait_async0();        // this wave's async writes for buffer (it&1) landed
    __syncthreads();      // all waves landed; prior readers of next buffer done
    if (it + 1 < niter) issue_tile(it + 1);   // overlap next loads with WMMAs

    const __bf16* Ab = As[it & 1];
    const __bf16* Bb = Bs[it & 1];

    // fragment loads (ISA 16-bit A layout: lanes<16 k{0..7,16..23},
    // lanes>=16 k{8..15,24..31}; B symmetric per-column)
    Frag a[4], bfr[2];
#pragma unroll
    for (int mi = 0; mi < 4; ++mi) {
      const __bf16* p = &Ab[(wm * 64 + mi * 16 + lrow) * LDSTR + koff];
      a[mi].u[0] = *(const u32x4*)p;
      a[mi].u[1] = *(const u32x4*)(p + 16);
    }
#pragma unroll
    for (int ni = 0; ni < 2; ++ni) {
      const __bf16* p = &Bb[(wn * 32 + ni * 16 + lrow) * LDSTR + koff];
      bfr[ni].u[0] = *(const u32x4*)p;
      bfr[ni].u[1] = *(const u32x4*)(p + 16);
    }

#pragma unroll
    for (int mi = 0; mi < 4; ++mi)
#pragma unroll
      for (int ni = 0; ni < 2; ++ni)
        acc[mi][ni] = __builtin_amdgcn_wmma_f32_16x16x32_bf16(
            false, a[mi].v, false, bfr[ni].v, (short)0, acc[mi][ni], false, false);
    __syncthreads();      // all reads of buffer (it&1) done before reuse
  }

  // ---- epilogue: + bias, store fp32 gates ----
  const int rofs = (lane < 16) ? 0 : 8;
#pragma unroll
  for (int ni = 0; ni < 2; ++ni) {
    const int col = bn + wn * 32 + ni * 16 + lrow;
    const float bv = bias[col];
#pragma unroll
    for (int mi = 0; mi < 4; ++mi) {
      const int rbase = bm + wm * 64 + mi * 16 + rofs;
#pragma unroll
      for (int j = 0; j < 8; ++j)
        outg[(long)(rbase + j) * k4H + col] = acc[mi][ni][j] + bv;
    }
  }
}

// ---------------------------------------------------------------------------
// LSTM pointwise: gate order i,f,g,o (torch). fp32 cell, bf16 hidden out.
// ---------------------------------------------------------------------------
__global__ __launch_bounds__(256)
void lstm_pointwise(const float* __restrict__ g, float* __restrict__ c,
                    __bf16* __restrict__ hb)
{
  const int idx = blockIdx.x * 256 + threadIdx.x;   // over B*H
  const int b   = idx >> 10;
  const int hh  = idx & (kH - 1);
  const float* gr = g + (long)b * k4H;
  const float i  = sigf(gr[hh]);
  const float f  = sigf(gr[kH + hh]);
  const float gg = tanhf(gr[2 * kH + hh]);
  const float o  = sigf(gr[3 * kH + hh]);
  const float cn = f * c[idx] + i * gg;
  c[idx]  = cn;
  hb[idx] = (__bf16)(o * tanhf(cn));
}

// logits[b,t,:] = h1[b,:] @ fcW^T + fcb   (N=37 is tiny; VALU dot is fine)
__global__ __launch_bounds__(256)
void fc_logits(const __bf16* __restrict__ h1, const float* __restrict__ W,
               const float* __restrict__ bias, float* __restrict__ out, int t)
{
  const int idx = blockIdx.x * 256 + threadIdx.x;
  if (idx >= kB * kV) return;
  const int b = idx / kV;
  const int v = idx - b * kV;
  const __bf16* hr = h1 + (long)b * kH;
  const float*  wr = W  + (long)v * kH;
  float acc = 0.f;
#pragma unroll 8
  for (int k = 0; k < kH; ++k) acc += (float)hr[k] * wr[k];
  out[((long)b * kT + t) * kV + v] = acc + bias[v];
}

// ---- one-time conversions ----
__global__ __launch_bounds__(256)
void build_wcat(const float* __restrict__ Wih, int K0,
                const float* __restrict__ Whh, int K1,
                __bf16* __restrict__ Wcat)
{
  const int  Kt  = K0 + K1;
  const long idx = (long)blockIdx.x * 256 + threadIdx.x;
  if (idx >= (long)k4H * Kt) return;
  const int n = (int)(idx / Kt);
  const int k = (int)(idx - (long)n * Kt);
  const float v = (k < K0) ? Wih[(long)n * K0 + k] : Whh[(long)n * K1 + (k - K0)];
  Wcat[idx] = (__bf16)v;
}

__global__ __launch_bounds__(256)
void f32_to_bf16(const float* __restrict__ a, __bf16* __restrict__ o, int n)
{
  const int i = blockIdx.x * 256 + threadIdx.x;
  if (i < n) o[i] = (__bf16)a[i];
}

__global__ __launch_bounds__(256)
void vec_add(const float* __restrict__ a, const float* __restrict__ b,
             float* __restrict__ o, int n)
{
  const int i = blockIdx.x * 256 + threadIdx.x;
  if (i < n) o[i] = a[i] + b[i];
}

__global__ __launch_bounds__(256)
void init_state(float* __restrict__ c0, float* __restrict__ c1,
                __bf16* __restrict__ h0, __bf16* __restrict__ h1)
{
  const int i = blockIdx.x * 256 + threadIdx.x;  // B*H
  c0[i] = 0.f; c1[i] = 0.f;
  h0[i] = (__bf16)0.f; h1[i] = (__bf16)0.f;
}

// ---------------------------------------------------------------------------
extern "C" void kernel_launch(void* const* d_in, const int* in_sizes, int n_in,
                              void* d_out, int out_size, void* d_ws, size_t ws_size,
                              hipStream_t stream)
{
  (void)in_sizes; (void)n_in; (void)out_size; (void)ws_size;
  const int*   src    = (const int*)  d_in[0];
  const int*   tgt    = (const int*)  d_in[1];
  const float* emb    = (const float*)d_in[2];
  const float* eW_ih0 = (const float*)d_in[3];
  const float* eW_hh0 = (const float*)d_in[4];
  const float* eb_ih0 = (const float*)d_in[5];
  const float* eb_hh0 = (const float*)d_in[6];
  const float* eW_ih1 = (const float*)d_in[7];
  const float* eW_hh1 = (const float*)d_in[8];
  const float* eb_ih1 = (const float*)d_in[9];
  const float* eb_hh1 = (const float*)d_in[10];
  const float* dW_ih0 = (const float*)d_in[11];
  const float* dW_hh0 = (const float*)d_in[12];
  const float* db_ih0 = (const float*)d_in[13];
  const float* db_hh0 = (const float*)d_in[14];
  const float* dW_ih1 = (const float*)d_in[15];
  const float* dW_hh1 = (const float*)d_in[16];
  const float* db_ih1 = (const float*)d_in[17];
  const float* db_hh1 = (const float*)d_in[18];
  const float* fcW    = (const float*)d_in[19];
  const float* fcb    = (const float*)d_in[20];
  float* out = (float*)d_out;

  // ---- carve workspace (~113 MB) ----
  char* ws = (char*)d_ws;
  size_t off = 0;
  auto carve = [&](size_t bytes) -> char* {
    char* p = ws + off;
    off += (bytes + 255) & ~(size_t)255;
    return p;
  };
  __bf16* emb_b = (__bf16*)carve((size_t)kV * kE * 2);
  __bf16* Wce0  = (__bf16*)carve((size_t)k4H * (kE + kH) * 2);   // [4096,1280]
  __bf16* Wce1  = (__bf16*)carve((size_t)k4H * (2 * kH) * 2);    // [4096,2048]
  __bf16* Wcd0  = (__bf16*)carve((size_t)k4H * (kE + kH) * 2);
  __bf16* Wcd1  = (__bf16*)carve((size_t)k4H * (2 * kH) * 2);
  float*  bs0   = (float*) carve((size_t)k4H * 4);
  float*  bs1   = (float*) carve((size_t)k4H * 4);
  float*  bs2   = (float*) carve((size_t)k4H * 4);
  float*  bs3   = (float*) carve((size_t)k4H * 4);
  float*  gates = (float*) carve((size_t)kB * k4H * 4);
  float*  c0    = (float*) carve((size_t)kB * kH * 4);
  float*  c1    = (float*) carve((size_t)kB * kH * 4);
  __bf16* h0b   = (__bf16*)carve((size_t)kB * kH * 2);
  __bf16* h1b   = (__bf16*)carve((size_t)kB * kH * 2);

  // ---- one-time conversions (deterministic every call) ----
  f32_to_bf16<<<(kV * kE + 255) / 256, 256, 0, stream>>>(emb, emb_b, kV * kE);
  build_wcat<<<(int)(((long)k4H * (kE + kH) + 255) / 256), 256, 0, stream>>>(eW_ih0, kE, eW_hh0, kH, Wce0);
  build_wcat<<<(int)(((long)k4H * (2 * kH)  + 255) / 256), 256, 0, stream>>>(eW_ih1, kH, eW_hh1, kH, Wce1);
  build_wcat<<<(int)(((long)k4H * (kE + kH) + 255) / 256), 256, 0, stream>>>(dW_ih0, kE, dW_hh0, kH, Wcd0);
  build_wcat<<<(int)(((long)k4H * (2 * kH)  + 255) / 256), 256, 0, stream>>>(dW_ih1, kH, dW_hh1, kH, Wcd1);
  vec_add<<<k4H / 256, 256, 0, stream>>>(eb_ih0, eb_hh0, bs0, k4H);
  vec_add<<<k4H / 256, 256, 0, stream>>>(eb_ih1, eb_hh1, bs1, k4H);
  vec_add<<<k4H / 256, 256, 0, stream>>>(db_ih0, db_hh0, bs2, k4H);
  vec_add<<<k4H / 256, 256, 0, stream>>>(db_ih1, db_hh1, bs3, k4H);
  init_state<<<kB * kH / 256, 256, 0, stream>>>(c0, c1, h0b, h1b);

  const dim3 ggrid(k4H / 128, kB / 128);  // 32 x 16 workgroups
  const int  pgrid = kB * kH / 256;

  // ---- encoder: 24 recurrent steps ----
  for (int t = 0; t < kS; ++t) {
    gates_gemm<<<ggrid, 256, 0, stream>>>(emb_b, kE, kE, h0b, kH,
                                          1, src + t, kS, 0,
                                          Wce0, bs0, gates);
    lstm_pointwise<<<pgrid, 256, 0, stream>>>(gates, c0, h0b);
    gates_gemm<<<ggrid, 256, 0, stream>>>(h0b, kH, kH, h1b, kH,
                                          0, nullptr, 0, 0,
                                          Wce1, bs1, gates);
    lstm_pointwise<<<pgrid, 256, 0, stream>>>(gates, c1, h1b);
  }

  // ---- decoder: 24 steps, teacher-forced tokens [SOS, tgt[:, :-1]] ----
  for (int t = 0; t < kT; ++t) {
    const int* tok = (t == 0) ? nullptr : (tgt + (t - 1));   // null -> constant SOS
    gates_gemm<<<ggrid, 256, 0, stream>>>(emb_b, kE, kE, h0b, kH,
                                          1, tok, kT, kSOS,
                                          Wcd0, bs2, gates);
    lstm_pointwise<<<pgrid, 256, 0, stream>>>(gates, c0, h0b);
    gates_gemm<<<ggrid, 256, 0, stream>>>(h0b, kH, kH, h1b, kH,
                                          0, nullptr, 0, 0,
                                          Wcd1, bs3, gates);
    lstm_pointwise<<<pgrid, 256, 0, stream>>>(gates, c1, h1b);
    fc_logits<<<(kB * kV + 255) / 256, 256, 0, stream>>>(h1b, fcW, fcb, out, t);
  }
}
